// SimplestTransformer_82188494176639
// MI455X (gfx1250) — matrix operand order repaired and checked by reading, not compiled, not run on hardware
//
#include <hip/hip_runtime.h>
#include <hip/hip_bf16.h>

// ---------------------------------------------------------------------------
// CDNA5 (gfx1250) bf16 WMMA implementation of the 8-layer causal-mean MLP LM.
// GEMM: TDM (tensor_load_to_lds) double-buffered tile staging -> ds_load_b128
// fragment loads -> v_wmma_f32_16x16x32_bf16 (f32 accumulate).
// Fallback staging: global_load_async_to_lds_b128 (proven to compile).
// ---------------------------------------------------------------------------

typedef __attribute__((ext_vector_type(16))) __bf16 v16bf;
typedef __attribute__((ext_vector_type(8)))  float  v8f;

#define TILE_M 128
#define TILE_N 128
#define TILE_K 32
#define LDS_STRIDE 40  // ushorts/row: 32 data (64B) + 8 pad (16B) -> TDM pad 4dw/16dw

#if defined(__gfx1250__) && \
    __has_builtin(__builtin_amdgcn_global_load_async_to_lds_b128) && \
    __has_builtin(__builtin_amdgcn_s_wait_asynccnt)
#define USE_ASYNC_COPY 1
typedef int v4i_vs __attribute__((vector_size(16)));
typedef __attribute__((address_space(1))) v4i_vs* as1_v4i_ptr;
typedef __attribute__((address_space(3))) v4i_vs* as3_v4i_ptr;
#else
#define USE_ASYNC_COPY 0
#endif

#if USE_ASYNC_COPY && \
    __has_builtin(__builtin_amdgcn_tensor_load_to_lds) && \
    __has_builtin(__builtin_amdgcn_s_wait_tensorcnt)
#define USE_TDM 1
typedef unsigned int tdm_v4u __attribute__((vector_size(16)));
typedef int          tdm_v8i __attribute__((vector_size(32)));
typedef int          tdm_v4i __attribute__((vector_size(16)));

__device__ __forceinline__ unsigned lds_addr_of(const void* p) {
    return (unsigned)(unsigned long long)(as3_v4i_ptr)p;   // addrspacecast + ptrtoint
}

// 2-D tile load via Tensor Data Mover.  data_size=2B.  LDS padding: 4 dwords
// after every 16 dwords (64B row -> 80B row, matches LDS_STRIDE=40 ushorts).
__device__ __forceinline__ void tdm_load_2d(unsigned long long gaddr, unsigned ldsaddr,
                                            unsigned tensor_d0, unsigned tensor_d1,
                                            unsigned tile_d0, unsigned tile_d1,
                                            unsigned long long stride0) {
    tdm_v4u g0;
    g0[0] = 1u;                                   // count=1, user descriptor
    g0[1] = ldsaddr;                              // lds_addr
    g0[2] = (unsigned)gaddr;                      // global_addr[31:0]
    g0[3] = (unsigned)(gaddr >> 32) | (2u << 30); // global_addr[56:32] | type=2
    tdm_v8i g1;
    g1[0] = (int)((1u << 16)      // data_size = 2 bytes
                | (1u << 20)      // pad_enable
                | (3u << 22)      // pad_interval: 16 dwords
                | (3u << 25));    // pad_amount:   4 dwords
    g1[1] = (int)(tensor_d0 << 16);                           // tensor_dim0[15:0]
    g1[2] = (int)((tensor_d0 >> 16) | (tensor_d1 << 16));     // dim0 hi | dim1 lo
    g1[3] = (int)((tensor_d1 >> 16) | (tile_d0 << 16));       // dim1 hi | tile_dim0
    g1[4] = (int)(tile_d1 & 0xFFFFu);                         // tile_dim1 (tile_dim2=0)
    g1[5] = (int)(unsigned)stride0;                           // tensor_dim0_stride[31:0]
    g1[6] = (int)(unsigned)(stride0 >> 32);                   // stride0[47:32] (dim1_stride=0)
    g1[7] = 0;
    tdm_v4i g2;
    g2[0] = 1;            // tensor_dim2 = 1
    g2[1] = 1;            // tensor_dim3 = 1 (iterate_enable=0)
    g2[2] = 0;            // tensor_dim2_stride lo
    g2[3] = 0;            // stride hi | tile_dim3 = 0
    tdm_v4i g3;
    g3[0] = 0;            // tensor_dim3_stride lo
    g3[1] = (int)(1u << 16);  // stride hi=0 | tensor_dim4 = 1
    g3[2] = 0;            // tensor_dim4 hi | tile_dim4 = 0
    g3[3] = 0;
    tdm_v8i g4 = {0, 0, 0, 0, 0, 0, 0, 0};   // 6-arg toolchain: extra group, zero
    __builtin_amdgcn_tensor_load_to_lds(g0, g1, g2, g3, g4, 0);
}
#else
#define USE_TDM 0
#endif

__device__ __forceinline__ unsigned short f32_to_bf16(float f) {
    unsigned int u = __builtin_bit_cast(unsigned int, f);
    u += 0x7FFFu + ((u >> 16) & 1u);   // round-to-nearest-even
    return (unsigned short)(u >> 16);
}
__device__ __forceinline__ unsigned int pack_bf16x2(float lo, float hi) {
    unsigned int a = __builtin_bit_cast(unsigned int, lo);
    unsigned int b = __builtin_bit_cast(unsigned int, hi);
    a += 0x7FFFu + ((a >> 16) & 1u);
    b += 0x7FFFu + ((b >> 16) & 1u);
    return (a >> 16) | (b & 0xFFFF0000u);
}

// ---------------------------------------------------------------------------
// GEMM: C[M,N] (f32) = A[M,K] (bf16 row-major) x B[N,K] (bf16 row-major,
// i.e. B(k,n) = Bsrc[n*K+k]) + optional bias.  Both operands identical layout.
// ---------------------------------------------------------------------------
__global__ __launch_bounds__(256, 2)
void gemm_bf16_wmma(const unsigned short* __restrict__ A,
                    const unsigned short* __restrict__ Bsrc,
                    const float* __restrict__ bias, float* __restrict__ C,
                    int M, int N, int K) {
    __shared__ __align__(16) unsigned short As[2][TILE_M * LDS_STRIDE];
    __shared__ __align__(16) unsigned short Bs[2][TILE_N * LDS_STRIDE];

    const int t    = threadIdx.x;
    const int lane = t & 31;
    const int lr   = lane & 15;     // row-within-fragment
    const int half = lane >> 4;     // 0: lanes 0-15, 1: lanes 16-31
    const int wave = t >> 5;        // 8 waves
    const int wm   = wave >> 2;     // 0..1 -> 64 rows each
    const int wn   = wave & 3;      // 0..3 -> 32 cols each

    const int m0 = blockIdx.y * TILE_M;
    const int n0 = blockIdx.x * TILE_N;
    const int nk = K / TILE_K;

    v8f acc[4][2] = {};

#if USE_TDM
    // ---- prologue: wave0 DMAs tile 0 into buffer 0 ----
    if (wave == 0) {
        tdm_load_2d((unsigned long long)(const void*)(A + (size_t)m0 * K),
                    lds_addr_of(&As[0][0]), K, M, TILE_K, TILE_M, K);
        tdm_load_2d((unsigned long long)(const void*)(Bsrc + (size_t)n0 * K),
                    lds_addr_of(&Bs[0][0]), K, N, TILE_K, TILE_N, K);
        __builtin_amdgcn_s_wait_tensorcnt(0);
    }
    __syncthreads();
#endif

    for (int kt = 0; kt < nk; ++kt) {
#if USE_TDM
        const int cur = kt & 1;
        if (wave == 0 && kt + 1 < nk) {        // prefetch next tile into other buffer
            int kk = (kt + 1) * TILE_K;
            tdm_load_2d((unsigned long long)(const void*)(A + (size_t)m0 * K + kk),
                        lds_addr_of(&As[cur ^ 1][0]), K, M, TILE_K, TILE_M, K);
            tdm_load_2d((unsigned long long)(const void*)(Bsrc + (size_t)n0 * K + kk),
                        lds_addr_of(&Bs[cur ^ 1][0]), K, N, TILE_K, TILE_N, K);
        }
#else
        const int cur = 0;
        const int k0 = kt * TILE_K;
#if USE_ASYNC_COPY
        #pragma unroll
        for (int j = 0; j < 2; ++j) {
            int u = t + 256 * j;       // 0..511
            int r = u >> 2;            // tile row 0..127
            int seg = (u & 3) * 8;     // bf16 element offset: 0,8,16,24
            const unsigned short* ga = A + (size_t)(m0 + r) * K + k0 + seg;
            const unsigned short* gb = Bsrc + (size_t)(n0 + r) * K + k0 + seg;
            __builtin_amdgcn_global_load_async_to_lds_b128(
                (as1_v4i_ptr)ga, (as3_v4i_ptr)&As[0][r * LDS_STRIDE + seg], 0, 0);
            __builtin_amdgcn_global_load_async_to_lds_b128(
                (as1_v4i_ptr)gb, (as3_v4i_ptr)&Bs[0][r * LDS_STRIDE + seg], 0, 0);
        }
        __builtin_amdgcn_s_wait_asynccnt(0);
#else
        #pragma unroll
        for (int j = 0; j < 2; ++j) {
            int u = t + 256 * j;
            int r = u >> 2;
            int seg = (u & 3) * 8;
            uint4 va = *(const uint4*)(A + (size_t)(m0 + r) * K + k0 + seg);
            uint4 vb = *(const uint4*)(Bsrc + (size_t)(n0 + r) * K + k0 + seg);
            *(uint4*)(&As[0][r * LDS_STRIDE + seg]) = va;
            *(uint4*)(&Bs[0][r * LDS_STRIDE + seg]) = vb;
        }
#endif
        __syncthreads();
#endif  // !USE_TDM

        // ---- fragment loads per ISA 7.12.2 VGPR layouts ----
        v16bf afr[4], bfr[2];
        #pragma unroll
        for (int mf = 0; mf < 4; ++mf) {
            const unsigned short* base = &As[cur][(wm * 64 + mf * 16 + lr) * LDS_STRIDE];
            union { v16bf v; unsigned int u[8]; } fa;
            #pragma unroll
            for (int i = 0; i < 8; ++i) {
                // A 16-bit 16x32: vgpr i<4 -> K = 8*half + 2i ; i>=4 -> 16 + 8*half + 2(i-4)
                int k = (i < 4) ? (8 * half + 2 * i) : (16 + 8 * half + 2 * (i - 4));
                fa.u[i] = *(const unsigned int*)(base + k);
            }
            afr[mf] = fa.v;
        }
        #pragma unroll
        for (int nf = 0; nf < 2; ++nf) {
            const unsigned short* base = &Bs[cur][(wn * 32 + nf * 16 + lr) * LDS_STRIDE];
            union { v16bf v; unsigned int u[8]; } fb;
            #pragma unroll
            for (int i = 0; i < 8; ++i) {
                // B 16-bit 32x16: vgpr i<4 -> K = 8*half + 2i ; i>=4 -> 16 + 8*half + 2(i-4)
                fb.u[i] = *(const unsigned int*)(base + ((i < 4) ? (8 * half + 2 * i) : (16 + 8 * half + 2 * (i - 4))));
            }
            bfr[nf] = fb.v;
        }

        #pragma unroll
        for (int mf = 0; mf < 4; ++mf)
            #pragma unroll
            for (int nf = 0; nf < 2; ++nf)
                acc[mf][nf] = __builtin_amdgcn_wmma_f32_16x16x32_bf16(
                    false, afr[mf], false, bfr[nf],
                    (short)0, acc[mf][nf], false, false);

#if USE_TDM
        if (wave == 0) __builtin_amdgcn_s_wait_tensorcnt(0);  // next buffer resident
#endif
        __syncthreads();
    }

    // ---- epilogue: C/D layout — lane col = lr, row = r + 8*half ----
    #pragma unroll
    for (int mf = 0; mf < 4; ++mf)
        #pragma unroll
        for (int nf = 0; nf < 2; ++nf) {
            int col = n0 + wn * 32 + nf * 16 + lr;
            float b = bias ? bias[col] : 0.0f;
            #pragma unroll
            for (int r = 0; r < 8; ++r) {
                int row = m0 + wm * 64 + mf * 16 + 8 * half + r;
                C[(size_t)row * N + col] = acc[mf][nf][r] + b;
            }
        }
}

// ---------------------------------------------------------------------------
// Prep / epilogue kernels (bandwidth-trivial vs the GEMMs)
// ---------------------------------------------------------------------------

__global__ void embed_gather(const int* __restrict__ seq, const float* __restrict__ emb,
                             float* __restrict__ x, unsigned short* __restrict__ xb,
                             int BS, int D) {
    long long e = (long long)(blockIdx.x * blockDim.x + threadIdx.x) * 4;
    if (e >= (long long)BS * D) return;
    int row = (int)(e / D), d = (int)(e % D);
    int tok = seq[row];
    float4 v = *(const float4*)(emb + (size_t)tok * D + d);
    *(float4*)(x + (size_t)row * D + d) = v;
    uint2 p;
    p.x = pack_bf16x2(v.x, v.y);
    p.y = pack_bf16x2(v.z, v.w);
    *(uint2*)(xb + (size_t)row * D + d) = p;
}

__global__ void convert_bf16(const float* __restrict__ src, unsigned short* __restrict__ dst,
                             long long n) {
    long long e = (long long)(blockIdx.x * blockDim.x + threadIdx.x) * 4;
    if (e >= n) return;
    float4 v = *(const float4*)(src + e);
    uint2 p;
    p.x = pack_bf16x2(v.x, v.y);
    p.y = pack_bf16x2(v.z, v.w);
    *(uint2*)(dst + e) = p;
}

// dst[c][r] = bf16(src[r][c]) : tiled transpose+convert (out_W [D,V] -> [V,D])
__global__ void transpose_convert_bf16(const float* __restrict__ src,
                                       unsigned short* __restrict__ dst,
                                       int R, int Cc) {
    __shared__ float tile[32][33];
    int c0 = blockIdx.x * 32, r0 = blockIdx.y * 32;
    int tx = threadIdx.x & 31, ty = threadIdx.x >> 5;   // 32x8 threads
    #pragma unroll
    for (int i = 0; i < 32; i += 8)
        tile[ty + i][tx] = src[(size_t)(r0 + ty + i) * Cc + c0 + tx];
    __syncthreads();
    #pragma unroll
    for (int i = 0; i < 32; i += 8)
        dst[(size_t)(c0 + ty + i) * R + r0 + tx] = f32_to_bf16(tile[tx][ty + i]);
}

// x[b,s,d] += relu(cumsum_s(v)/(s+1));  also refresh xb = bf16(x)
__global__ void causal_mean_relu_add(const float* __restrict__ v, float* __restrict__ x,
                                     unsigned short* __restrict__ xb, int B, int S, int D) {
    int t = blockIdx.x * blockDim.x + threadIdx.x;
    if (t >= B * D) return;
    int b = t / D, d = t % D;
    const float*    vp  = v  + (size_t)b * S * D + d;
    float*          xp  = x  + (size_t)b * S * D + d;
    unsigned short* xbp = xb + (size_t)b * S * D + d;
    float sum = 0.0f;
    for (int s = 0; s < S; ++s) {
        sum += vp[(size_t)s * D];
        float nx = xp[(size_t)s * D] + fmaxf(sum / (float)(s + 1), 0.0f);
        xp[(size_t)s * D]  = nx;
        xbp[(size_t)s * D] = f32_to_bf16(nx);
    }
}

extern "C" void kernel_launch(void* const* d_in, const int* in_sizes, int n_in,
                              void* d_out, int out_size, void* d_ws, size_t ws_size,
                              hipStream_t stream) {
    const int*   seq  = (const int*)d_in[0];
    const float* emb  = (const float*)d_in[1];
    const float* Ws   = (const float*)d_in[2];
    const float* outW = (const float*)d_in[3];
    const float* outb = (const float*)d_in[4];
    float*       out  = (float*)d_out;

    const int B = 4, S = 2048, D = 1024, V = 32000, L = 8;
    const int BS = B * S;                       // 8192

    float*          x   = (float*)d_ws;                         // 32 MB
    float*          v   = x + (size_t)BS * D;                   // 32 MB
    unsigned short* xb  = (unsigned short*)(v + (size_t)BS * D);// 16 MB
    unsigned short* Wb  = xb + (size_t)BS * D;                  // 16 MB (8 x D x D)
    unsigned short* Wtb = Wb + (size_t)L * D * D;               // 64 MB (V x D)

    // weight prep (once per launch)
    {
        long long nW = (long long)L * D * D;                    // 8M
        convert_bf16<<<(int)(nW / 4 / 256), 256, 0, stream>>>(Ws, Wb, nW);
        dim3 g(V / 32, D / 32);                                 // (1000, 32)
        transpose_convert_bf16<<<g, 256, 0, stream>>>(outW, Wtb, D, V);
    }
    // embedding gather (f32 + bf16)
    embed_gather<<<BS * D / 4 / 256, 256, 0, stream>>>(seq, emb, x, xb, BS, D);

    // layers
    for (int i = 0; i < L; ++i) {
        dim3 grid(D / TILE_N, BS / TILE_M);                     // (8, 64)
        gemm_bf16_wmma<<<grid, 256, 0, stream>>>(xb, Wb + (size_t)i * D * D,
                                                 nullptr, v, BS, D, D);
        causal_mean_relu_add<<<(B * D + 255) / 256, 256, 0, stream>>>(v, x, xb, B, S, D);
    }
    // vocab projection with fused bias
    {
        dim3 grid(V / TILE_N, BS / TILE_M);                     // (250, 64)
        gemm_bf16_wmma<<<grid, 256, 0, stream>>>(xb, Wtb, outb, out, BS, V, D);
    }
}
